// VTBlock_71614284693900
// MI455X (gfx1250) — matrix operand order repaired and checked by reading, not compiled, not run on hardware
//
#include <hip/hip_runtime.h>
#include <hip/hip_bf16.h>
#include <math.h>

typedef __attribute__((ext_vector_type(16))) __bf16 v16bf;
typedef __attribute__((ext_vector_type(8)))  float  v8f;

#define Pq 57600   // 240*240 pixels
#define Dq 256
#define Nq 240
#define Hq 8
#define LWq 20
#define Fq 400

__device__ __forceinline__ unsigned short f2bf(float x) {
  union { float f; unsigned int u; } a; a.f = x;
  unsigned int u = a.u;
  if ((u & 0x7F800000u) == 0x7F800000u) return (unsigned short)(u >> 16); // inf/nan: truncate
  unsigned int r = u + 0x7FFFu + ((u >> 16) & 1u);                        // RNE
  return (unsigned short)(r >> 16);
}

__global__ void zero_kernel(float* __restrict__ x, int n) {
  int i = blockIdx.x * blockDim.x + threadIdx.x;
  if (i < n) x[i] = 0.f;
}

__global__ void cvt_bf16_kernel(const float* __restrict__ x, unsigned short* __restrict__ y, int n) {
  int i = blockIdx.x * blockDim.x + threadIdx.x;
  if (i < n) y[i] = f2bf(x[i]);
}

// LayerNorm over channel dim (C=256), input (C,P) f32 -> output (C,P) bf16
__global__ void ln_kernel(const float* __restrict__ x, const float* __restrict__ g,
                          const float* __restrict__ b, unsigned short* __restrict__ y) {
  int p = blockIdx.x * blockDim.x + threadIdx.x;
  if (p >= Pq) return;
  float s = 0.f;
  for (int c = 0; c < Dq; ++c) s += x[(size_t)c * Pq + p];
  float mean = s * (1.f / Dq);
  float v = 0.f;
  for (int c = 0; c < Dq; ++c) { float d = x[(size_t)c * Pq + p] - mean; v += d * d; }
  float rinv = rsqrtf(v * (1.f / Dq) + 1e-5f);
  for (int c = 0; c < Dq; ++c) {
    float xn = (x[(size_t)c * Pq + p] - mean) * rinv;
    y[(size_t)c * Pq + p] = f2bf(xn * g[c] + b[c]);
  }
}

// Generic GEMM: per block 16M x 64N, 4 waves each own a 16x16 N-subtile.
// B panel (32K x 64N bf16 = 4KB) is staged to LDS with async global->LDS B128
// copies (ASYNCcnt), then all waves build B fragments from LDS.
// mode 0: +bias, scale rows<256 by qscale, store bf16   (qkv)
// mode 1: +bias, +resid(f32), store f32                 (out-proj + residual)
// mode 2: +bias, ReLU, store bf16                       (lin1)
// mode 3: accumulate += into outf (f32)                 (lin2 chunk)
#define BSTRIDE 80  // LDS row stride in u16 (64 data + 16 pad -> 160B, 16B aligned, bank spread)
__global__ __launch_bounds__(128) void gemm_bf16_kernel(
    const unsigned short* __restrict__ A, int lda,
    const unsigned short* __restrict__ B, int K,
    const float* __restrict__ bias, const float* __restrict__ resid,
    float* __restrict__ outf, unsigned short* __restrict__ outbf,
    int mode, float qscale) {
  __shared__ unsigned short Blds[32 * BSTRIDE];
  int tid  = threadIdx.x;
  int lane = tid & 31;
  int wid  = tid >> 5;
  int m0   = blockIdx.y * 16;
  int nblk = blockIdx.x * 64;
  int n0   = nblk + wid * 16;
  int half = lane >> 4;
  int mrow = lane & 15;
  int ncol = lane & 15;

  // LDS byte offset of the panel (low 32 bits of a shared-aperture pointer)
  unsigned ldsbase = (unsigned)(size_t)(&Blds[0]);

  const unsigned short* Arow = A + (size_t)(m0 + mrow) * lda;
  v8f acc = {};
  for (int k0 = 0; k0 < K; k0 += 32) {
    // ---- async stage B panel: 32 rows x 128B, 256 chunks of 16B ----
    {
      int c0 = tid;          // chunks tid and tid+128
#pragma unroll
      for (int it = 0; it < 2; ++it) {
        int c   = c0 + it * 128;
        int row = c >> 3;
        int col = (c & 7) * 8;                       // element offset in row
        const unsigned short* gsrc = B + (size_t)(k0 + row) * Pq + nblk + col;
        unsigned long long ga = (unsigned long long)(size_t)gsrc;
        unsigned ldst = ldsbase + (unsigned)(row * (BSTRIDE * 2) + col * 2);
        asm volatile("global_load_async_to_lds_b128 %0, %1, off"
                     :: "v"(ldst), "v"(ga) : "memory");
      }
      __builtin_prefetch(B + (size_t)(k0 + 32) * Pq + nblk, 0, 1);
      asm volatile("s_wait_asynccnt 0x0" ::: "memory");
    }
    __syncthreads();

    // ---- A fragment: two contiguous 16B global loads (ISA 16-bit A layout) ----
    union { v16bf v; uint4 q[2]; } ua;
    ua.q[0] = *(const uint4*)(Arow + k0 + half * 8);
    ua.q[1] = *(const uint4*)(Arow + k0 + 16 + half * 8);

    // ---- B fragment from LDS: column gather, ISA 16-bit layout (K 8h..8h+7, 16+8h..16+8h+7) ----
    union { v16bf v; unsigned short s[16]; } ub;
#pragma unroll
    for (int t = 0; t < 16; ++t)
      ub.s[t] = Blds[((t < 8) ? (half * 8 + t) : (16 + half * 8 + (t - 8))) * BSTRIDE + wid * 16 + ncol];

    acc = __builtin_amdgcn_wmma_f32_16x16x32_bf16(false, ua.v, false, ub.v,
                                                  (short)0, acc, false, false);
    __syncthreads();  // panel reuse barrier before next overwrite
  }

  union { v8f v; float f[8]; } uc; uc.v = acc;
  int n = n0 + ncol;
#pragma unroll
  for (int r = 0; r < 8; ++r) {
    int m = m0 + r + half * 8;
    size_t idx = (size_t)m * Pq + n;
    float val = uc.f[r];
    if (mode == 0)      { val += bias[m]; if (m < 256) val *= qscale; outbf[idx] = f2bf(val); }
    else if (mode == 1) { val += bias[m]; val += resid[idx]; outf[idx] = val; }
    else if (mode == 2) { val += bias[m]; val = val > 0.f ? val : 0.f; outbf[idx] = f2bf(val); }
    else                { outf[idx] += val; }
  }
}

__device__ __forceinline__ int pix_of(int y0, int x0, int i) {
  return (y0 + i / LWq) * Nq + (x0 + i % LWq);
}

// One wave per (window, head, 16-row tile). qkv: (768 x P) bf16. out: (256 x P) bf16.
__global__ __launch_bounds__(32) void attn_kernel(
    const unsigned short* __restrict__ qkv, const float* __restrict__ relb,
    const unsigned char* __restrict__ pm, unsigned short* __restrict__ outbf) {
  __shared__ float S[16 * Fq];
  __shared__ int s_full;
  int lane = threadIdx.x;
  int blk = blockIdx.x;
  int mt = blk % 25;
  int h  = (blk / 25) % Hq;
  int w  = blk / 200;
  int y0 = (w / 12) * LWq, x0 = (w % 12) * LWq;

  const unsigned short* q  = qkv + (size_t)(h * 32) * Pq;
  const unsigned short* kk = qkv + (size_t)(256 + h * 32) * Pq;
  const unsigned short* vv = qkv + (size_t)(512 + h * 32) * Pq;

  // fully-padded window detection (mask reset path)
  if (lane == 0) s_full = 1;
  __syncthreads();
  int allp = 1;
  for (int j = lane; j < Fq; j += 32)
    if (!pm[pix_of(y0, x0, j)]) allp = 0;
  if (!allp) atomicAnd(&s_full, 0);
  __syncthreads();
  int full = s_full;

  int half = lane >> 4;
  int mrow = lane & 15;
  int ncol = lane & 15;
  int i0 = mt * 16;
  int pi = pix_of(y0, x0, i0 + mrow);

  // A fragment: rows of q^T (K = hd = 32, one WMMA covers it)
  union { v16bf v; unsigned short s[16]; } ua;
#pragma unroll
  for (int c = 0; c < 16; ++c) {
    int kd = (c < 8) ? (half * 8 + c) : (16 + half * 8 + (c - 8));
    ua.s[c] = q[(size_t)kd * Pq + pi];
  }

  // scores: 25 column tiles, epilogue applies key-padding mask + rel bias -> LDS
  for (int jt = 0; jt < 25; ++jt) {
    int j0 = jt * 16;
    int jcol = j0 + ncol;
    int pj = pix_of(y0, x0, jcol);
    union { v16bf v; unsigned short s[16]; } ub;
#pragma unroll
    for (int t = 0; t < 16; ++t) ub.s[t] = kk[(size_t)((t < 8) ? (half * 8 + t) : (16 + half * 8 + (t - 8))) * Pq + pj];
    v8f c = {};
    c = __builtin_amdgcn_wmma_f32_16x16x32_bf16(false, ua.v, false, ub.v,
                                                (short)0, c, false, false);
    int colpad = (!full) && pm[pj];
    int lyj = jcol / LWq, lxj = jcol % LWq;
    union { v8f v; float f[8]; } uc; uc.v = c;
#pragma unroll
    for (int r = 0; r < 8; ++r) {
      int m = r + half * 8;
      int i = i0 + m;
      int lyi = i / LWq, lxi = i % LWq;
      float val = colpad ? -__builtin_inff() : uc.f[r];
      int r0 = (lyi - lyj + 39) % 39;
      int r1 = (lxi - lxj + 39) % 39;
      val += relb[(r0 * 39 + r1) * Hq + h];
      S[m * Fq + jcol] = val;
    }
  }
  __syncthreads();

  // softmax: one lane per row
  if (lane < 16) {
    float mx = -__builtin_inff();
    for (int j = 0; j < Fq; ++j) mx = fmaxf(mx, S[lane * Fq + j]);
    float sum = 0.f;
    for (int j = 0; j < Fq; ++j) { float e = __expf(S[lane * Fq + j] - mx); S[lane * Fq + j] = e; sum += e; }
    float inv = 1.f / sum;
    for (int j = 0; j < Fq; ++j) S[lane * Fq + j] *= inv;
  }
  __syncthreads();

  // O = attn(16xF) * V^T(F x 32): K=400 = 12*32 + 16 (zero-padded tail), two N tiles (hd=32)
  v8f o0 = {}, o1 = {};
  for (int kt = 0; kt < 13; ++kt) {
    int jb = kt * 32;
    union { v16bf v; unsigned short s[16]; } sa;
#pragma unroll
    for (int c = 0; c < 16; ++c) {
      int kd = (c < 8) ? (half * 8 + c) : (16 + half * 8 + (c - 8));
      int j = jb + kd;
      sa.s[c] = (j < Fq) ? f2bf(S[mrow * Fq + j]) : (unsigned short)0;
    }
    union { v16bf v; unsigned short s[16]; } vb0, vb1;
#pragma unroll
    for (int t = 0; t < 16; ++t) {
      int j = jb + ((t < 8) ? (half * 8 + t) : (16 + half * 8 + (t - 8)));
      if (j < Fq) {
        int pj = pix_of(y0, x0, j);
        vb0.s[t] = vv[(size_t)(ncol) * Pq + pj];
        vb1.s[t] = vv[(size_t)(16 + ncol) * Pq + pj];
      } else { vb0.s[t] = 0; vb1.s[t] = 0; }
    }
    o0 = __builtin_amdgcn_wmma_f32_16x16x32_bf16(false, sa.v, false, vb0.v, (short)0, o0, false, false);
    o1 = __builtin_amdgcn_wmma_f32_16x16x32_bf16(false, sa.v, false, vb1.v, (short)0, o1, false, false);
  }
  union { v8f v; float f[8]; } u0, u1; u0.v = o0; u1.v = o1;
#pragma unroll
  for (int r = 0; r < 8; ++r) {
    int m = r + half * 8;
    int p = pix_of(y0, x0, i0 + m);
    outbf[(size_t)(h * 32 + ncol) * Pq + p]      = f2bf(u0.f[r]);
    outbf[(size_t)(h * 32 + 16 + ncol) * Pq + p] = f2bf(u1.f[r]);
  }
}

// out = mask ? 0 : src1 + ffn_acc + lin2_b
__global__ void final_kernel(const float* __restrict__ src1, float* __restrict__ out,
                             const float* __restrict__ b2, const unsigned char* __restrict__ pm, int n) {
  int i = blockIdx.x * blockDim.x + threadIdx.x;
  if (i >= n) return;
  int c = i / Pq;
  int p = i - c * Pq;
  out[i] = pm[p] ? 0.f : (src1[i] + out[i] + b2[c]);
}

extern "C" void kernel_launch(void* const* d_in, const int* in_sizes, int n_in,
                              void* d_out, int out_size, void* d_ws, size_t ws_size,
                              hipStream_t stream) {
  const float*         src    = (const float*)d_in[0];
  const unsigned char* pmask  = (const unsigned char*)d_in[1];
  const float* qkv_w  = (const float*)d_in[2];
  const float* qkv_b  = (const float*)d_in[3];
  const float* out_w  = (const float*)d_in[4];
  const float* out_b  = (const float*)d_in[5];
  const float* lin1_w = (const float*)d_in[6];
  const float* lin1_b = (const float*)d_in[7];
  const float* lin2_w = (const float*)d_in[8];
  const float* lin2_b = (const float*)d_in[9];
  const float* n1_g   = (const float*)d_in[10];
  const float* n1_b   = (const float*)d_in[11];
  const float* n2_g   = (const float*)d_in[12];
  const float* n2_b   = (const float*)d_in[13];
  const float* relb   = (const float*)d_in[14];
  float* out = (float*)d_out;

  char* ws = (char*)d_ws;
  unsigned short* qkvw_bf = (unsigned short*)(ws + 0);          // 768*256
  unsigned short* outw_bf = (unsigned short*)(ws + 393216);     // 256*256
  unsigned short* l1w_bf  = (unsigned short*)(ws + 524288);     // 2048*256
  unsigned short* l2w_bf  = (unsigned short*)(ws + 1572864);    // 256*2048
  unsigned short* regA    = (unsigned short*)(ws + 2621440);    // 256*P bf16 (xn / attnout / ln2)
  unsigned short* regB    = (unsigned short*)(ws + 32112640);   // 768*P bf16 (qkv / ffn hidden chunk)
  float*          regC    = (float*)(ws + 120586240);           // 256*P f32 (src + attn residual)

  const int total = Dq * Pq;
  const float qscale = 0.17677669529663689f; // 1/sqrt(32)

  zero_kernel<<<(total + 255) / 256, 256, 0, stream>>>(out, total);
  cvt_bf16_kernel<<<(768 * 256 + 255) / 256, 256, 0, stream>>>(qkv_w, qkvw_bf, 768 * 256);
  cvt_bf16_kernel<<<(256 * 256 + 255) / 256, 256, 0, stream>>>(out_w, outw_bf, 256 * 256);
  cvt_bf16_kernel<<<(2048 * 256 + 255) / 256, 256, 0, stream>>>(lin1_w, l1w_bf, 2048 * 256);
  cvt_bf16_kernel<<<(256 * 2048 + 255) / 256, 256, 0, stream>>>(lin2_w, l2w_bf, 256 * 2048);

  // LN1 -> xn (bf16)
  ln_kernel<<<(Pq + 255) / 256, 256, 0, stream>>>(src, n1_g, n1_b, regA);
  // qkv GEMM (q rows pre-scaled)
  gemm_bf16_kernel<<<dim3(Pq / 64, 768 / 16), 128, 0, stream>>>(
      qkvw_bf, 256, regA, 256, qkv_b, nullptr, nullptr, regB, 0, qscale);
  // windowed attention -> attnout (bf16, regA)
  attn_kernel<<<144 * Hq * 25, 32, 0, stream>>>(regB, relb, pmask, regA);
  // out-proj + residual -> src1 (f32, regC)
  gemm_bf16_kernel<<<dim3(Pq / 64, 256 / 16), 128, 0, stream>>>(
      outw_bf, 256, regA, 256, out_b, src, regC, nullptr, 1, 0.f);
  // LN2 -> bf16 (regA)
  ln_kernel<<<(Pq + 255) / 256, 256, 0, stream>>>(regC, n2_g, n2_b, regA);
  // FFN in 4 chunks of 512 DFF channels; lin2 accumulates into d_out (f32)
  for (int ch = 0; ch < 4; ++ch) {
    gemm_bf16_kernel<<<dim3(Pq / 64, 512 / 16), 128, 0, stream>>>(
        l1w_bf + (size_t)ch * 512 * 256, 256, regA, 256,
        lin1_b + ch * 512, nullptr, nullptr, regB, 2, 0.f);
    gemm_bf16_kernel<<<dim3(Pq / 64, 256 / 16), 128, 0, stream>>>(
        l2w_bf + (size_t)ch * 512, 2048, regB, 512,
        nullptr, nullptr, out, nullptr, 3, 0.f);
  }
  // residual + lin2 bias + padding zero
  final_kernel<<<(total + 255) / 256, 256, 0, stream>>>(regC, out, lin2_b, pmask, total);
}
